// MoEGroupGEMM_80169859547412
// MI455X (gfx1250) — compile-verified, matrix operand order checked by reading, not run on hardware
//
#include <hip/hip_runtime.h>
#include <hip/hip_bf16.h>
#include <cstdint>

#define NUM_TOKENS 2048
#define DIM 1024
#define NUM_EXPERTS 8
#define BM 128
#define BN 128
#define BK 64
#define LDK 80  // padded LDS row stride (elements): 160B, multiple of 16B

typedef __attribute__((ext_vector_type(16))) __bf16 v16bf;
typedef __attribute__((ext_vector_type(8)))  float  v8f;
typedef __attribute__((ext_vector_type(4)))  unsigned int vuint4;

// ---- CDNA5 async global->LDS copy (ASYNCcnt-tracked), 16B per lane ----------
__device__ __forceinline__ void async_b128(__bf16* lds_ptr, const __bf16* gptr) {
  unsigned lds = (unsigned)(uintptr_t)lds_ptr;  // generic ptr low 32 = LDS addr
  asm volatile("global_load_async_to_lds_b128 %0, %1, off"
               :: "v"(lds), "v"(gptr) : "memory");
}
__device__ __forceinline__ void wait_async0() {
  asm volatile("s_wait_asynccnt 0x0" ::: "memory");
}

// Build a 16x16x32 bf16 fragment from an LDS row (fragment-major rows).
__device__ __forceinline__ v16bf frag_from_lds(const __bf16* rowbase, int ks, int hl) {
  v16bf r;
  const __bf16* p0 = rowbase + ks + hl * 8;
  const __bf16* p1 = rowbase + ks + 16 + hl * 8;
#pragma unroll
  for (int i = 0; i < 8; ++i) { r[i] = p0[i]; r[i + 8] = p1[i]; }
  return r;
}

__global__ void k_init(int* counts) {
  if (threadIdx.x < NUM_EXPERTS) counts[threadIdx.x] = 0;
}

__global__ __launch_bounds__(256) void k_cvt_bf16(const float* __restrict__ src,
                                                  __bf16* __restrict__ dst, int n) {
  int i = (blockIdx.x * 256 + threadIdx.x) * 4;
  if (i + 3 < n) {
    float4 v = *(const float4*)(src + i);
    dst[i + 0] = (__bf16)v.x;
    dst[i + 1] = (__bf16)v.y;
    dst[i + 2] = (__bf16)v.z;
    dst[i + 3] = (__bf16)v.w;
  }
}

// Convert + transpose one expert weight: dst[e][n][k] = (bf16) src[e][k][n].
// 64x64 tiles via LDS; grid = e(8) x kt(16) x nt(16) flattened.
__global__ __launch_bounds__(256) void k_cvt_transpose(const float* __restrict__ src,
                                                       __bf16* __restrict__ dst) {
  int b = blockIdx.x;
  int e = b >> 8;
  int kt = (b >> 4) & 15;
  int nt = b & 15;
  __shared__ __bf16 t[64][65];
  const float* s = src + ((size_t)e << 20) + (size_t)(kt * 64) * DIM + nt * 64;
  int tx = threadIdx.x & 63;  // n within tile (coalesced reads)
  int ty = threadIdx.x >> 6;  // 0..3
#pragma unroll
  for (int i = 0; i < 16; ++i) {
    int k = ty * 16 + i;
    t[tx][k] = (__bf16)s[(size_t)k * DIM + tx];
  }
  __syncthreads();
#pragma unroll
  for (int it = 0; it < 2; ++it) {
    int c = threadIdx.x + it * 256;  // 512 chunks of 8 bf16
    int n = c >> 3;
    int kc = (c & 7) * 8;
    __align__(16) __bf16 tmp[8];
#pragma unroll
    for (int j = 0; j < 8; ++j) tmp[j] = t[n][kc + j];
    __bf16* d = dst + ((size_t)e << 20) + (size_t)(nt * 64 + n) * DIM + kt * 64 + kc;
    *(vuint4*)d = *(const vuint4*)tmp;
  }
}

// Router: logits -> softmax -> top2 (lowest index wins ties, matching lax.top_k).
__global__ __launch_bounds__(256) void k_router(const float* __restrict__ tokens,
                                                const float* __restrict__ rw,
                                                int* __restrict__ counts,
                                                int* __restrict__ eids,
                                                float* __restrict__ wgts) {
  __shared__ float lw[NUM_EXPERTS * DIM];  // 32 KB
  for (int i = threadIdx.x; i < NUM_EXPERTS * DIM; i += 256) lw[i] = rw[i];
  __syncthreads();

  int t = blockIdx.x * 256 + threadIdx.x;
  const float* tr = tokens + (size_t)t * DIM;
  float acc[NUM_EXPERTS];
#pragma unroll
  for (int e = 0; e < NUM_EXPERTS; ++e) acc[e] = 0.f;
  for (int d = 0; d < DIM; ++d) {
    float x = tr[d];
#pragma unroll
    for (int e = 0; e < NUM_EXPERTS; ++e) acc[e] += x * lw[e * DIM + d];
  }
  float m = acc[0];
#pragma unroll
  for (int e = 1; e < NUM_EXPERTS; ++e) m = fmaxf(m, acc[e]);
  float p[NUM_EXPERTS];
  float s = 0.f;
#pragma unroll
  for (int e = 0; e < NUM_EXPERTS; ++e) { p[e] = expf(acc[e] - m); s += p[e]; }
  float inv = 1.0f / s;
#pragma unroll
  for (int e = 0; e < NUM_EXPERTS; ++e) p[e] *= inv;

  int e0 = 0; float b0 = p[0];
#pragma unroll
  for (int e = 1; e < NUM_EXPERTS; ++e) if (p[e] > b0) { b0 = p[e]; e0 = e; }
  int e1 = -1; float b1 = -1.f;
#pragma unroll
  for (int e = 0; e < NUM_EXPERTS; ++e)
    if (e != e0 && p[e] > b1) { b1 = p[e]; e1 = e; }

  eids[t * 2 + 0] = e0;
  eids[t * 2 + 1] = e1;
  wgts[t * 2 + 0] = b0;
  wgts[t * 2 + 1] = b1;
  atomicAdd(counts + e0, 1);
  atomicAdd(counts + e1, 1);
}

__global__ void k_scan(const int* __restrict__ counts, int* __restrict__ offsets,
                       int* __restrict__ cursors) {
  if (threadIdx.x == 0) {
    int off = 0;
    for (int e = 0; e < NUM_EXPERTS; ++e) {
      offsets[e] = off;
      cursors[e] = off;
      off += counts[e];
    }
  }
}

__global__ __launch_bounds__(256) void k_assign(const int* __restrict__ eids,
                                                int* __restrict__ cursors,
                                                int* __restrict__ rows,
                                                int* __restrict__ slots) {
  int t = blockIdx.x * 256 + threadIdx.x;
#pragma unroll
  for (int k = 0; k < 2; ++k) {
    int e = eids[t * 2 + k];
    int pslot = atomicAdd(cursors + e, 1);
    rows[pslot] = t;
    slots[t * 2 + k] = pslot;
  }
}

// Grouped GEMM1: H[slot,:] = gelu_exact( gather(tok)[slot,:] @ W1[e] ), bf16 out.
// B operand comes pre-transposed (w1t[e][n][k]); both tiles staged by async
// global->LDS copies with double buffering.
__global__ __launch_bounds__(256) void k_gemm1(const __bf16* __restrict__ tokbf,
                                               const __bf16* __restrict__ w1t,
                                               const int* __restrict__ rows,
                                               const int* __restrict__ counts,
                                               const int* __restrict__ offsets,
                                               __bf16* __restrict__ Hbuf) {
  const int e  = blockIdx.x >> 4;
  const int mt = blockIdx.x & 15;
  const int nb = blockIdx.y;
  const int cnt = counts[e];
  if (mt * BM >= cnt) return;  // block-uniform: EXEC stays full for WMMA
  const int seg = offsets[e];

  __shared__ __align__(16) __bf16 ldsA[2][BM * LDK];
  __shared__ __align__(16) __bf16 ldsB[2][BN * LDK];

  const int tid  = threadIdx.x;
  const int lane = tid & 31;
  const int wid  = tid >> 5;
  const int wm = (wid & 3) * 32;   // wave grid 4(M) x 2(N): 32x64 per wave
  const int wn = (wid >> 2) * 64;
  const int hl = lane >> 4;
  const int lq = lane & 15;

  // Per-thread chunk plan: 1024 16B chunks per tile, 4 per thread; A and B
  // share the same (row, kc) pattern. Gather indices clamped in-range
  // (garbage rows only feed accumulator rows >= cnt, which are never stored).
  const __bf16* arow[4];
  const __bf16* brow[4];
  int loff[4];
#pragma unroll
  for (int it = 0; it < 4; ++it) {
    int c = tid + it * 256;
    int r = c >> 3;
    int kc = (c & 7) * 8;
    loff[it] = r * LDK + kc;
    int gm = mt * BM + r;
    int idx = seg + (gm < cnt ? gm : cnt - 1);
    arow[it] = tokbf + (size_t)rows[idx] * DIM + kc;
    brow[it] = w1t + ((size_t)e << 20) + (size_t)(nb * BN + r) * DIM + kc;
  }

  v8f acc[2][4];
#pragma unroll
  for (int mi = 0; mi < 2; ++mi)
#pragma unroll
    for (int ni = 0; ni < 4; ++ni)
      acc[mi][ni] = (v8f){0.f, 0.f, 0.f, 0.f, 0.f, 0.f, 0.f, 0.f};

  // Prologue: fill buffer 0.
#pragma unroll
  for (int it = 0; it < 4; ++it) {
    async_b128(&ldsA[0][loff[it]], arow[it]);
    async_b128(&ldsB[0][loff[it]], brow[it]);
  }
  wait_async0();
  __syncthreads();

  for (int kb = 0; kb < DIM; kb += BK) {
    const int buf = (kb / BK) & 1;
    if (kb + BK < DIM) {  // prefetch next tile into the other buffer
#pragma unroll
      for (int it = 0; it < 4; ++it) {
        async_b128(&ldsA[buf ^ 1][loff[it]], arow[it] + kb + BK);
        async_b128(&ldsB[buf ^ 1][loff[it]], brow[it] + kb + BK);
      }
    }
#pragma unroll
    for (int ks = 0; ks < BK; ks += 32) {
      v16bf af[2], bfr[4];
#pragma unroll
      for (int mi = 0; mi < 2; ++mi)
        af[mi] = frag_from_lds(&ldsA[buf][(wm + mi * 16 + lq) * LDK], ks, hl);
#pragma unroll
      for (int ni = 0; ni < 4; ++ni)
        bfr[ni] = frag_from_lds(&ldsB[buf][(wn + ni * 16 + lq) * LDK], ks, hl);
#pragma unroll
      for (int mi = 0; mi < 2; ++mi)
#pragma unroll
        for (int ni = 0; ni < 4; ++ni)
          acc[mi][ni] = __builtin_amdgcn_wmma_f32_16x16x32_bf16(
              false, af[mi], false, bfr[ni], (short)0, acc[mi][ni], false, false);
    }
    wait_async0();   // next buffer resident
    __syncthreads(); // all reads of current buffer retired (dscnt waited)
  }

  // Epilogue: exact GELU, store bf16 H (C layout: VGPR j -> row j+8*hl, col lq).
#pragma unroll
  for (int mi = 0; mi < 2; ++mi)
#pragma unroll
    for (int ni = 0; ni < 4; ++ni)
#pragma unroll
      for (int j = 0; j < 8; ++j) {
        int gm = mt * BM + wm + mi * 16 + hl * 8 + j;
        if (gm < cnt) {
          float x = acc[mi][ni][j];
          float g = 0.5f * x * (1.0f + erff(x * 0.70710678118654752440f));
          int col = nb * BN + wn + ni * 16 + lq;
          Hbuf[(size_t)(seg + gm) * DIM + col] = (__bf16)g;
        }
      }
}

// Grouped GEMM2: Y[slot,:] = H[slot,:] @ W2[e] (w2 pre-transposed), f32 out.
__global__ __launch_bounds__(256) void k_gemm2(const __bf16* __restrict__ Hbuf,
                                               const __bf16* __restrict__ w2t,
                                               const int* __restrict__ counts,
                                               const int* __restrict__ offsets,
                                               float* __restrict__ Ybuf) {
  const int e  = blockIdx.x >> 4;
  const int mt = blockIdx.x & 15;
  const int nb = blockIdx.y;
  const int cnt = counts[e];
  if (mt * BM >= cnt) return;
  const int seg = offsets[e];

  __shared__ __align__(16) __bf16 ldsA[2][BM * LDK];
  __shared__ __align__(16) __bf16 ldsB[2][BN * LDK];

  const int tid  = threadIdx.x;
  const int lane = tid & 31;
  const int wid  = tid >> 5;
  const int wm = (wid & 3) * 32;
  const int wn = (wid >> 2) * 64;
  const int hl = lane >> 4;
  const int lq = lane & 15;

  const __bf16* arow[4];
  const __bf16* brow[4];
  int loff[4];
#pragma unroll
  for (int it = 0; it < 4; ++it) {
    int c = tid + it * 256;
    int r = c >> 3;
    int kc = (c & 7) * 8;
    loff[it] = r * LDK + kc;
    int gm = mt * BM + r;
    int idx = seg + (gm < cnt ? gm : cnt - 1);
    arow[it] = Hbuf + (size_t)idx * DIM + kc;
    brow[it] = w2t + ((size_t)e << 20) + (size_t)(nb * BN + r) * DIM + kc;
  }

  v8f acc[2][4];
#pragma unroll
  for (int mi = 0; mi < 2; ++mi)
#pragma unroll
    for (int ni = 0; ni < 4; ++ni)
      acc[mi][ni] = (v8f){0.f, 0.f, 0.f, 0.f, 0.f, 0.f, 0.f, 0.f};

#pragma unroll
  for (int it = 0; it < 4; ++it) {
    async_b128(&ldsA[0][loff[it]], arow[it]);
    async_b128(&ldsB[0][loff[it]], brow[it]);
  }
  wait_async0();
  __syncthreads();

  for (int kb = 0; kb < DIM; kb += BK) {
    const int buf = (kb / BK) & 1;
    if (kb + BK < DIM) {
#pragma unroll
      for (int it = 0; it < 4; ++it) {
        async_b128(&ldsA[buf ^ 1][loff[it]], arow[it] + kb + BK);
        async_b128(&ldsB[buf ^ 1][loff[it]], brow[it] + kb + BK);
      }
    }
#pragma unroll
    for (int ks = 0; ks < BK; ks += 32) {
      v16bf af[2], bfr[4];
#pragma unroll
      for (int mi = 0; mi < 2; ++mi)
        af[mi] = frag_from_lds(&ldsA[buf][(wm + mi * 16 + lq) * LDK], ks, hl);
#pragma unroll
      for (int ni = 0; ni < 4; ++ni)
        bfr[ni] = frag_from_lds(&ldsB[buf][(wn + ni * 16 + lq) * LDK], ks, hl);
#pragma unroll
      for (int mi = 0; mi < 2; ++mi)
#pragma unroll
        for (int ni = 0; ni < 4; ++ni)
          acc[mi][ni] = __builtin_amdgcn_wmma_f32_16x16x32_bf16(
              false, af[mi], false, bfr[ni], (short)0, acc[mi][ni], false, false);
    }
    wait_async0();
    __syncthreads();
  }

#pragma unroll
  for (int mi = 0; mi < 2; ++mi)
#pragma unroll
    for (int ni = 0; ni < 4; ++ni)
#pragma unroll
      for (int j = 0; j < 8; ++j) {
        int gm = mt * BM + wm + mi * 16 + hl * 8 + j;
        if (gm < cnt) {
          int col = nb * BN + wn + ni * 16 + lq;
          Ybuf[(size_t)(seg + gm) * DIM + col] = acc[mi][ni][j];
        }
      }
}

// out[t,:] = w0*Y[slot0,:] + w1*Y[slot1,:]  (fixed 2-term sum -> deterministic)
__global__ __launch_bounds__(256) void k_combine(const float* __restrict__ Ybuf,
                                                 const int* __restrict__ slots,
                                                 const float* __restrict__ wgts,
                                                 float* __restrict__ out) {
  int t = blockIdx.x;
  int s0 = slots[t * 2 + 0], s1 = slots[t * 2 + 1];
  float w0 = wgts[t * 2 + 0], w1 = wgts[t * 2 + 1];
  const float* y0 = Ybuf + (size_t)s0 * DIM;
  const float* y1 = Ybuf + (size_t)s1 * DIM;
  float* o = out + (size_t)t * DIM;
  for (int n = threadIdx.x; n < DIM; n += 256) o[n] = w0 * y0[n] + w1 * y1[n];
}

extern "C" void kernel_launch(void* const* d_in, const int* in_sizes, int n_in,
                              void* d_out, int out_size, void* d_ws, size_t ws_size,
                              hipStream_t stream) {
  (void)in_sizes; (void)n_in; (void)out_size; (void)ws_size;
  const float* tokens   = (const float*)d_in[0];
  const float* router_w = (const float*)d_in[1];
  const float* w1       = (const float*)d_in[2];
  const float* w2       = (const float*)d_in[3];
  float* out = (float*)d_out;

  char* base = (char*)d_ws;
  size_t off = 0;
  auto carve = [&](size_t bytes) -> char* {
    char* p = base + off;
    off += (bytes + 255) & ~(size_t)255;
    return p;
  };
  __bf16* tokbf = (__bf16*)carve((size_t)NUM_TOKENS * DIM * 2);         //  4 MiB
  __bf16* w1t   = (__bf16*)carve((size_t)NUM_EXPERTS * DIM * DIM * 2);  // 16 MiB (transposed)
  __bf16* w2t   = (__bf16*)carve((size_t)NUM_EXPERTS * DIM * DIM * 2);  // 16 MiB (transposed)
  __bf16* Hbuf  = (__bf16*)carve((size_t)2 * NUM_TOKENS * DIM * 2);     //  8 MiB
  float*  Ybuf  = (float*)carve((size_t)2 * NUM_TOKENS * DIM * 4);      // 16 MiB
  int*    eids  = (int*)carve(NUM_TOKENS * 2 * sizeof(int));
  float*  wgts  = (float*)carve(NUM_TOKENS * 2 * sizeof(float));
  int*    slots = (int*)carve(NUM_TOKENS * 2 * sizeof(int));
  int*    rows  = (int*)carve(NUM_TOKENS * 2 * sizeof(int));
  int*    counts = (int*)carve(64);
  int*    offs   = (int*)carve(64);
  int*    curs   = (int*)carve(64);

  k_init<<<1, 32, 0, stream>>>(counts);
  k_cvt_bf16<<<(NUM_TOKENS * DIM) / 1024, 256, 0, stream>>>(tokens, tokbf, NUM_TOKENS * DIM);
  k_cvt_transpose<<<NUM_EXPERTS * 16 * 16, 256, 0, stream>>>(w1, w1t);
  k_cvt_transpose<<<NUM_EXPERTS * 16 * 16, 256, 0, stream>>>(w2, w2t);
  k_router<<<NUM_TOKENS / 256, 256, 0, stream>>>(tokens, router_w, counts, eids, wgts);
  k_scan<<<1, 1, 0, stream>>>(counts, offs, curs);
  k_assign<<<NUM_TOKENS / 256, 256, 0, stream>>>(eids, curs, rows, slots);
  k_gemm1<<<dim3(NUM_EXPERTS * 16, DIM / BN), 256, 0, stream>>>(tokbf, w1t, rows, counts, offs, Hbuf);
  k_gemm2<<<dim3(NUM_EXPERTS * 16, DIM / BN), 256, 0, stream>>>(Hbuf, w2t, counts, offs, Ybuf);
  k_combine<<<NUM_TOKENS, 256, 0, stream>>>(Ybuf, slots, wgts, out);
}
